// RepBNNBlock_88991722373191
// MI455X (gfx1250) — 8-bit matrix operand order repaired and checked by reading, compiled, not run on hardware
//
#include <hip/hip_runtime.h>
#include <hip/hip_bf16.h>

typedef __attribute__((ext_vector_type(16))) int   v16i;
typedef __attribute__((ext_vector_type(8)))  float v8f;
typedef int vint4 __attribute__((vector_size(16)));

#define NPIX   100352
#define NBLK_R 112
#define PIX_PER_BLK 896

#define GLOBAL_AS __attribute__((address_space(1)))
#define LDS_AS    __attribute__((address_space(3)))

__global__ __launch_bounds__(256) void bn_partial(const float* __restrict__ x,
                                                  float* __restrict__ psum,
                                                  float* __restrict__ psum2) {
    __shared__ float s1[256], s2[256];
    const int tid  = threadIdx.x;
    const int c    = tid & 127;
    const int half = tid >> 7;
    const long p0  = (long)blockIdx.x * PIX_PER_BLK;
    float sum = 0.f, sum2 = 0.f;
    for (int i = half; i < PIX_PER_BLK; i += 2) {
        float v = x[(p0 + i) * 128 + c];
        sum += v; sum2 += v * v;
    }
    s1[tid] = sum; s2[tid] = sum2;
    __syncthreads();
    if (tid < 128) {
        psum [blockIdx.x * 128 + c] = s1[tid] + s1[tid + 128];
        psum2[blockIdx.x * 128 + c] = s2[tid] + s2[tid + 128];
    }
}

__global__ void bn_final(const float* __restrict__ psum, const float* __restrict__ psum2,
                         float* __restrict__ meanArr, float* __restrict__ invArr) {
    int c = threadIdx.x;
    float s = 0.f, s2 = 0.f;
    for (int b = 0; b < NBLK_R; ++b) { s += psum[b * 128 + c]; s2 += psum2[b * 128 + c]; }
    float mean = s / (float)NPIX;
    float var  = s2 / (float)NPIX - mean * mean;
    meanArr[c] = mean;
    invArr[c]  = rsqrtf(var + 1e-3f);
}

__global__ void wscale(const float* __restrict__ W, float* __restrict__ scaleArr) {
    __shared__ float s[128];
    const int co = blockIdx.x, tid = threadIdx.x;
    float a = 0.f;
    for (int e = tid; e < 1152; e += 128) a += fabsf(W[(size_t)e * 128 + co]);
    s[tid] = a; __syncthreads();
    for (int off = 64; off; off >>= 1) { if (tid < off) s[tid] += s[tid + off]; __syncthreads(); }
    if (tid == 0) scaleArr[co] = s[0] / 1152.0f;
}

__global__ __launch_bounds__(256) void wpack_k(const float* __restrict__ W,
                                               unsigned char* __restrict__ wp) {
    int idx  = blockIdx.x * 256 + threadIdx.x;
    int j    = idx & 3;
    int v    = (idx >> 2) & 15;
    int lane = (idx >> 6) & 31;
    int nt   = (idx >> 11) & 7;
    int t    = idx >> 14;
    int ci = ((v >> 1) * 16) + ((lane >> 4) * 8) + ((v & 1) * 4) + j;
    int co = nt * 16 + (lane & 15);
    float w = W[((size_t)t * 128 + ci) * 128 + co];
    wp[idx] = (w >= 0.f) ? 0x38 : 0xB8;
}

__global__ __launch_bounds__(256) void xpack_k(const float* __restrict__ x,
                                               const float* __restrict__ gamma,
                                               const float* __restrict__ beta,
                                               const float* __restrict__ meanArr,
                                               const float* __restrict__ invArr,
                                               unsigned int* __restrict__ xp) {
    int idx = blockIdx.x * 256 + threadIdx.x;
    if (idx >= 3460608) return;
    int c4  = idx & 31;
    int pix = idx >> 5;
    int wp_ = pix % 58;
    int t2  = pix / 58;
    int hp  = t2 % 58;
    int b   = t2 / 58;
    unsigned int outw;
    if (hp == 0 || hp == 57 || wp_ == 0 || wp_ == 57) {
        outw = 0x38383838u;
    } else {
        const float4 xv = *(const float4*)(x +
            ((size_t)((b * 56 + hp - 1) * 56 + (wp_ - 1)) * 128 + c4 * 4));
        float vals[4] = {xv.x, xv.y, xv.z, xv.w};
        int c = c4 * 4;
        outw = 0u;
        #pragma unroll
        for (int k = 0; k < 4; ++k) {
            float bn = gamma[c + k] * (vals[k] - meanArr[c + k]) * invArr[c + k] + beta[c + k];
            outw |= ((bn >= 0.f) ? 0x38u : 0xB8u) << (8 * k);
        }
    }
    xp[idx] = outw;
}

__global__ __launch_bounds__(256) void conv_wmma(const unsigned char* __restrict__ xp,
                                                 unsigned char* __restrict__ wpack,
                                                 const float* __restrict__ scaleArr,
                                                 float* __restrict__ out) {
    extern __shared__ unsigned char smem[];
    const int tid = threadIdx.x;

#if defined(__has_builtin) && __has_builtin(__builtin_amdgcn_global_load_async_to_lds_b128)
    {
        GLOBAL_AS vint4* gsrc = (GLOBAL_AS vint4*)wpack;
        LDS_AS vint4*    ldst = (LDS_AS vint4*)smem;
        #pragma unroll 4
        for (int i = tid; i < 147456 / 16; i += 256) {
            __builtin_amdgcn_global_load_async_to_lds_b128(gsrc + i, ldst + i, 0, 0);
        }
        asm volatile("s_wait_asynccnt 0x0" ::: "memory");
    }
#else
    {
        const int4* src = (const int4*)wpack;
        int4*       dst = (int4*)smem;
        #pragma unroll 4
        for (int i = tid; i < 147456 / 16; i += 256) dst[i] = src[i];
    }
#endif
    __syncthreads();

    const int wave = tid >> 5;
    const int lane = tid & 31;
    const int lo   = lane & 15;
    const int hi   = lane >> 4;
    const int ntg  = wave & 3;
    const int mg   = wave >> 2;
    const int pixbase = blockIdx.x * 128 + mg * 64;
    const int abase   = hi * 8;

    size_t arow[4];
    #pragma unroll
    for (int m = 0; m < 4; ++m) {
        int pa = pixbase + m * 16 + lo;
        int b  = pa / 3136;
        int r  = pa - b * 3136;
        int h  = r / 56;
        int w  = r - h * 56;
        arow[m] = (size_t)((b * 58 + h) * 58 + w) * 128 + abase;
    }

    v8f zero = {0.f, 0.f, 0.f, 0.f, 0.f, 0.f, 0.f, 0.f};
    v8f acc[4][2];
    #pragma unroll
    for (int m = 0; m < 4; ++m) { acc[m][0] = zero; acc[m][1] = zero; }

    for (int t = 0; t < 9; ++t) {
        const int kh = t / 3, kw = t - kh * 3;
        const size_t tapoff = (size_t)(kh * 58 + kw) * 128;

        v16i B[2];
        #pragma unroll
        for (int n = 0; n < 2; ++n) {
            const int nt = ntg * 2 + n;
            const int4* bp = (const int4*)(smem + (((t * 8 + nt) * 32 + lane) << 6));
            #pragma unroll
            for (int q = 0; q < 4; ++q) {
                int4 d = bp[q];
                B[n][4 * q] = d.x; B[n][4 * q + 1] = d.y;
                B[n][4 * q + 2] = d.z; B[n][4 * q + 3] = d.w;
            }
        }

        #pragma unroll
        for (int m = 0; m < 4; ++m) {
            const unsigned char* ap = xp + arow[m] + tapoff;
            v16i A;
            #pragma unroll
            for (int i = 0; i < 8; ++i) {
                unsigned long long d = *(const unsigned long long*)(ap + 16 * i);
                A[2 * i]     = (int)(unsigned)(d);
                A[2 * i + 1] = (int)(unsigned)(d >> 32);
            }
            acc[m][0] = __builtin_amdgcn_wmma_f32_16x16x128_fp8_fp8(
                A, B[0], (short)0, acc[m][0], false, false);
            acc[m][1] = __builtin_amdgcn_wmma_f32_16x16x128_fp8_fp8(
                A, B[1], (short)0, acc[m][1], false, false);
        }
    }

    #pragma unroll
    for (int n = 0; n < 2; ++n) {
        const int   co = (ntg * 2 + n) * 16 + lo;
        const float s  = scaleArr[co];
        #pragma unroll
        for (int m = 0; m < 4; ++m) {
            const int prow = pixbase + m * 16 + 8 * hi;
            #pragma unroll
            for (int r = 0; r < 8; ++r) {
                out[(size_t)(prow + r) * 128 + co] = acc[m][n][r] * s;
            }
        }
    }
}

extern "C" void kernel_launch(void* const* d_in, const int* in_sizes, int n_in,
                              void* d_out, int out_size, void* d_ws, size_t ws_size,
                              hipStream_t stream) {
    const float* x     = (const float*)d_in[0];
    const float* gamma = (const float*)d_in[1];
    const float* beta  = (const float*)d_in[2];
    const float* W     = (const float*)d_in[3];
    float*       out   = (float*)d_out;
    char*        ws    = (char*)d_ws;

    float* meanArr  = (float*)(ws);
    float* invArr   = (float*)(ws + 512);
    float* scaleArr = (float*)(ws + 1024);
    float* psum     = (float*)(ws + 1536);
    float* psum2    = (float*)(ws + 58880);
    unsigned char* wp = (unsigned char*)(ws + 116224);
    unsigned char* xp = (unsigned char*)(ws + 263680);

    bn_partial<<<NBLK_R, 256, 0, stream>>>(x, psum, psum2);
    bn_final  <<<1, 128, 0, stream>>>(psum, psum2, meanArr, invArr);
    wscale    <<<128, 128, 0, stream>>>(W, scaleArr);
    wpack_k   <<<576, 256, 0, stream>>>(W, wp);
    xpack_k   <<<13518, 256, 0, stream>>>(x, gamma, beta, meanArr, invArr,
                                          (unsigned int*)xp);
    conv_wmma <<<784, 256, 147456, stream>>>(xp, wp, scaleArr, out);
}
